// BiAttention_44590350467841
// MI455X (gfx1250) — matrix operand order repaired and checked by reading, not compiled, not run on hardware
//
#include <hip/hip_runtime.h>
#include <hip/hip_bf16.h>

typedef __attribute__((ext_vector_type(16))) __bf16 v16bf;
typedef __attribute__((ext_vector_type(8)))  __bf16 v8bf;
typedef __attribute__((ext_vector_type(8)))  float  v8f;

#define B_    32
#define TC_   2048
#define TQ_   256
#define D_    256
#define VNEG  (-1e29f)
#define WAVES 8
#define ROWS_PER_WG (WAVES * 16)

// ---------------------------------------------------------------------------
// Pass 1: fused S = C*Q^T -> mask -> softmax (in place) -> c2q = P*Q.
// grid = (B, TC/128); block = 256 threads = 8 waves; each wave owns 16 rows.
// LDS (320KB, the CDNA5 per-WG limit):
//   Qbf [q][d] bf16 128KB  -> B operand of GEMM1 (contiguous K reads)
//   Qt  [d][q] bf16 128KB  -> B operand of GEMM2 (contiguous K reads)
//   S/P [wave][16][256] bf16 64KB -> scores, overwritten in place by P
// ---------------------------------------------------------------------------
__global__ void __launch_bounds__(WAVES * 32)
biattn_pass1(const float* __restrict__ ctx, const float* __restrict__ qry,
             const int* __restrict__ clen_a, const int* __restrict__ qlen_a,
             float* __restrict__ out1, float* __restrict__ mrow)
{
    extern __shared__ char smem[];
    __bf16* Qbf  = (__bf16*)smem;                       // Q row-major
    __bf16* Qt   = (__bf16*)(smem + 131072);            // Q transposed
    __bf16* Sall = (__bf16*)(smem + 262144);            // per-wave score tiles

    const int b    = blockIdx.x;
    const int tid  = threadIdx.x;
    const int wave = tid >> 5;
    const int lane = tid & 31;
    const int h    = lane >> 4;   // half-wave select per WMMA VGPR layout
    const int rl   = lane & 15;   // row (A) / column (B,C,D) within tile
    const int c0   = blockIdx.y * ROWS_PER_WG + wave * 16;
    const int clen = clen_a[b];
    const int qlen = qlen_a[b];

    // ---- stage question both row-major and transposed, f32 -> bf16 ----
    const float* qb = qry + (size_t)b * TQ_ * D_;
    for (int i = tid; i < TQ_ * D_; i += WAVES * 32) {
        const __bf16 bv = (__bf16)qb[i];          // coalesced global read
        Qbf[i] = bv;
        const int q = i >> 8, d = i & 255;
        Qt[d * TQ_ + q] = bv;                     // one-time scattered write
    }
    __syncthreads();

    __bf16* Sw = Sall + (size_t)wave * 16 * TQ_;

    // ---- A fragments: 16 context rows x 256 bf16. ISA 16-bit A layout:
    //      lane half h: K = 8h+e , 16+8h+e within each 32-wide K chunk. ----
    v16bf afr[8];
    {
        const float* crow = ctx + ((size_t)b * TC_ + c0 + rl) * D_;
#pragma unroll
        for (int kc = 0; kc < 8; ++kc) {
            const int k0 = kc * 32;
#pragma unroll
            for (int e = 0; e < 8; ++e) afr[kc][e]     = (__bf16)crow[k0 + 8*h + e];
#pragma unroll
            for (int e = 0; e < 8; ++e) afr[kc][8 + e] = (__bf16)crow[k0 + 16 + 8*h + e];
        }
    }

    // ---- GEMM1: S = C * Q^T ; B[k][n] = Q[n][k] -> contiguous row of Qbf ----
    for (int qt = 0; qt < 16; ++qt) {
        v8f acc = {0.f, 0.f, 0.f, 0.f, 0.f, 0.f, 0.f, 0.f};
        const __bf16* qrow = Qbf + (size_t)(qt * 16 + rl) * D_;
#pragma unroll
        for (int kc = 0; kc < 8; ++kc) {
            const int k0 = kc * 32;
            v16bf bfr;
#pragma unroll
            for (int e = 0; e < 16; ++e) bfr[e] = qrow[k0 + 8*h + (e & 7) + ((e >> 3) << 4)];
            acc = __builtin_amdgcn_wmma_f32_16x16x32_bf16(
                false, afr[kc], false, bfr, (short)0, acc, false, false);
        }
        // mask + spill bf16 score tile: C/D layout M = v + 8h, N = rl
        const int  qcol = qt * 16 + rl;
        const bool qok  = qcol < qlen;
#pragma unroll
        for (int v = 0; v < 8; ++v) {
            const int  m  = v + 8 * h;
            const bool ok = qok && ((c0 + m) < clen);
            Sw[m * TQ_ + qcol] = (__bf16)(ok ? acc[v] : VNEG);
        }
    }

    // ---- softmax: lane pair (rl, h) splits row rl into 128-wide halves;
    //      partials combined with cross-lane xor-16 shuffle. All row traffic
    //      in 8xbf16 chunks (ds_load_b128/ds_store_b128; bases 256B-aligned).
    //      exp written unnormalized in place, then scaled -> single exp pass. ----
    {
        v8bf* Sr = (v8bf*)(Sw + rl * TQ_ + h * 128);   // 16 chunks of 8
        float mx = VNEG;
        for (int cch = 0; cch < 16; ++cch) {
            const v8bf c = Sr[cch];
#pragma unroll
            for (int e = 0; e < 8; ++e) mx = fmaxf(mx, (float)c[e]);
        }
        mx = fmaxf(mx, __shfl_xor(mx, 16, 32));
        if (h == 0) mrow[(size_t)b * TC_ + c0 + rl] = mx;   // for q2c pass
        float sum = 0.f;
        for (int cch = 0; cch < 16; ++cch) {
            const v8bf c = Sr[cch];
            v8bf o;
#pragma unroll
            for (int e = 0; e < 8; ++e) {
                const float ev = __expf((float)c[e] - mx);
                sum += ev;
                o[e] = (__bf16)ev;
            }
            Sr[cch] = o;                                   // in-place, same addr
        }
        sum += __shfl_xor(sum, 16, 32);
        const float inv = 1.0f / sum;
        for (int cch = 0; cch < 16; ++cch) {
            const v8bf c = Sr[cch];
            v8bf o;
#pragma unroll
            for (int e = 0; e < 8; ++e) o[e] = (__bf16)((float)c[e] * inv);
            Sr[cch] = o;
        }
    }
    // same-wave DS ops are in order; EXEC fully re-converged before WMMA

    // ---- P fragments (A layout, bf16, contiguous rows of S/P region) ----
    v16bf pfr[8];
    {
        const __bf16* prow = Sw + (size_t)rl * TQ_;
#pragma unroll
        for (int kc = 0; kc < 8; ++kc) {
            const int k0 = kc * 32;
#pragma unroll
            for (int e = 0; e < 8; ++e) pfr[kc][e]     = prow[k0 + 8*h + e];
#pragma unroll
            for (int e = 0; e < 8; ++e) pfr[kc][8 + e] = prow[k0 + 16 + 8*h + e];
        }
    }

    // ---- GEMM2: O = P * Q ; B[k][n] = Q[k][n] = Qt[n][k] -> contiguous ----
    for (int nt = 0; nt < 16; ++nt) {
        v8f acc = {0.f, 0.f, 0.f, 0.f, 0.f, 0.f, 0.f, 0.f};
        const int ncol = nt * 16 + rl;
        const __bf16* qcolp = Qt + (size_t)ncol * TQ_;
#pragma unroll
        for (int kc = 0; kc < 8; ++kc) {
            const int k0 = kc * 32;
            v16bf bfr;
#pragma unroll
            for (int e = 0; e < 16; ++e) bfr[e] = qcolp[k0 + 8*h + (e & 7) + ((e >> 3) << 4)];
            acc = __builtin_amdgcn_wmma_f32_16x16x32_bf16(
                false, pfr[kc], false, bfr, (short)0, acc, false, false);
        }
#pragma unroll
        for (int v = 0; v < 8; ++v) {
            const int m = v + 8 * h;
            out1[((size_t)b * TC_ + c0 + m) * D_ + ncol] = acc[v];
        }
    }
}

// ---------------------------------------------------------------------------
// Pass 2: q2c = softmax_c(rowmax) . C, broadcast to [TC, D].
// grid = B; block = 256 (thread tid owns feature column d = tid).
// Context (64MB total) is L2-resident (192MB L2) after pass 1.
// ---------------------------------------------------------------------------
__global__ void __launch_bounds__(256)
biattn_pass2(const float* __restrict__ ctx, const float* __restrict__ mrow,
             float* __restrict__ out2)
{
    __shared__ float w[TC_];
    __shared__ float red[256];
    const int b   = blockIdx.x;
    const int tid = threadIdx.x;

    float lmax = VNEG;
    for (int c = tid; c < TC_; c += 256) {
        const float v = mrow[(size_t)b * TC_ + c];
        w[c] = v; lmax = fmaxf(lmax, v);
    }
    red[tid] = lmax; __syncthreads();
    for (int s = 128; s > 0; s >>= 1) {
        if (tid < s) red[tid] = fmaxf(red[tid], red[tid + s]);
        __syncthreads();
    }
    const float gmax = red[0]; __syncthreads();

    float lsum = 0.f;
    for (int c = tid; c < TC_; c += 256) {
        const float e = __expf(w[c] - gmax);
        w[c] = e; lsum += e;
    }
    red[tid] = lsum; __syncthreads();
    for (int s = 128; s > 0; s >>= 1) {
        if (tid < s) red[tid] += red[tid + s];
        __syncthreads();
    }
    const float inv = 1.0f / red[0]; __syncthreads();

    // weighted sum over all context rows for feature column tid (coalesced)
    float acc = 0.f;
    const float* cb = ctx + (size_t)b * TC_ * D_ + tid;
    for (int c = 0; c < TC_; ++c) acc += w[c] * cb[(size_t)c * D_];
    acc *= inv;

    float* ob = out2 + (size_t)b * TC_ * D_ + tid;
    for (int c = 0; c < TC_; ++c) ob[(size_t)c * D_] = acc;
}

// ---------------------------------------------------------------------------
extern "C" void kernel_launch(void* const* d_in, const int* in_sizes, int n_in,
                              void* d_out, int out_size, void* d_ws, size_t ws_size,
                              hipStream_t stream)
{
    const float* ctx  = (const float*)d_in[0];
    const float* qry  = (const float*)d_in[1];
    const int*   clen = (const int*)d_in[2];
    const int*   qlen = (const int*)d_in[3];

    float* out1 = (float*)d_out;
    float* out2 = out1 + (size_t)B_ * TC_ * D_;
    float* mrow = (float*)d_ws;   // B*TC f32 = 256KB scratch

    const size_t shmem = (size_t)TQ_ * D_ * 2          // Qbf  128KB
                       + (size_t)TQ_ * D_ * 2          // Qt   128KB
                       + (size_t)WAVES * 16 * TQ_ * 2; // S/P   64KB -> 320KB
    dim3 g1(B_, TC_ / ROWS_PER_WG);
    biattn_pass1<<<g1, WAVES * 32, shmem, stream>>>(ctx, qry, clen, qlen, out1, mrow);
    biattn_pass2<<<B_, 256, 0, stream>>>(ctx, mrow, out2);
}
